// ProteinMPNN_68332929679430
// MI455X (gfx1250) — matrix operand order repaired and checked by reading, not compiled, not run on hardware
//
#include <hip/hip_runtime.h>
#include <hip/hip_bf16.h>

// ---------------------------------------------------------------------------
// ProteinMPNN edge featurization + edge MLP projection for MI455X (gfx1250).
// B=4, L=2048, K=32 neighbors, 416-dim edge features, 128-dim output, LN.
// GEMM runs on v_wmma_f32_16x16x32_f16 (f32 accumulate), features built in LDS.
// ---------------------------------------------------------------------------

#define BB   4
#define LL   2048
#define KNB  32
#define NF   416          // 16 pos + 16 RBF(Ca-Ca) + 24*16 pair RBFs
#define NOUT 128
#define ROWS_PER_BLK 64   // edges per k_edge block
#define NROWS (BB*LL*KNB) // 262144 total edge rows

typedef __attribute__((ext_vector_type(16))) _Float16 v16h;
typedef __attribute__((ext_vector_type(8)))  _Float16 v8h;
typedef __attribute__((ext_vector_type(8)))  float    v8f;

__device__ __constant__ int cPA[24] = {0,2,3,4,1,1,1,1,0,0,0,4,4,3,0,2,3,4,2,3,4,2,3,2};
__device__ __constant__ int cPB[24] = {0,2,3,4,0,2,3,4,2,3,4,2,3,2,1,1,1,1,0,0,0,4,4,3};

__device__ __forceinline__ float rbf_val(float D, int m) {
    // mu = linspace(2,22,16); sigma = 20/16 = 1.25
    float mu = 2.0f + (20.0f / 15.0f) * (float)m;
    float t  = (D - mu) * 0.8f;
    return __expf(-t * t);
}

// ---------------------------------------------------------------------------
// Kernel 1: backbone atoms + virtual Cb -> ws atoms[(b*L+i)*15 + a*3 + d]
// ---------------------------------------------------------------------------
__global__ void k_atoms(const float* __restrict__ X, float* __restrict__ atoms) {
    int idx = blockIdx.x * 256 + threadIdx.x;
    if (idx >= BB * LL) return;
    const float* x = X + (size_t)idx * 12;
    float N[3], Ca[3], C[3], O[3], bb[3], cc[3], a[3], Cb[3];
    for (int d = 0; d < 3; d++) { N[d]=x[d]; Ca[d]=x[3+d]; C[d]=x[6+d]; O[d]=x[9+d]; }
    for (int d = 0; d < 3; d++) { bb[d] = Ca[d] - N[d]; cc[d] = C[d] - Ca[d]; }
    a[0] = bb[1]*cc[2] - bb[2]*cc[1];
    a[1] = bb[2]*cc[0] - bb[0]*cc[2];
    a[2] = bb[0]*cc[1] - bb[1]*cc[0];
    for (int d = 0; d < 3; d++)
        Cb[d] = -0.58273431f*a[d] + 0.56802827f*bb[d] - 0.54067466f*cc[d] + Ca[d];
    float* o = atoms + (size_t)idx * 15;
    for (int d = 0; d < 3; d++) { o[d]=N[d]; o[3+d]=Ca[d]; o[6+d]=C[d]; o[9+d]=O[d]; o[12+d]=Cb[d]; }
}

// ---------------------------------------------------------------------------
// Kernel 2: W_edge (416x128 f32 row-major) -> f16 column-major Wc[n*416+k]
// (B fragment reads one column per lane: K +{0..7,16..23} or +{8..15,24..31} by lane half)
// ---------------------------------------------------------------------------
__global__ void k_wstage(const float* __restrict__ We, _Float16* __restrict__ Wc) {
    int idx = blockIdx.x * 256 + threadIdx.x;      // 416*128 = 53248 = 208*256
    if (idx >= NF * NOUT) return;
    int n = idx / NF, k = idx - n * NF;
    Wc[idx] = (_Float16)We[k * NOUT + n];
}

// ---------------------------------------------------------------------------
// Kernel 3: masked top-32 nearest neighbors per residue (stable, ties->lower j)
// One block (256 threads) per (b,i). Writes eidx(int), dnb(D_adjust), float idx.
// ---------------------------------------------------------------------------
__global__ void k_topk(const float* __restrict__ atoms, const float* __restrict__ mask,
                       int* __restrict__ eidx, float* __restrict__ dnb,
                       float* __restrict__ out_eidx_f) {
    __shared__ float Dl[LL];
    __shared__ float red[256];
    __shared__ int   redi[256];
    const int tid  = threadIdx.x;
    const int bi   = blockIdx.x;          // b*L + i
    const int base = (bi >> 11) << 11;    // b*L
    const float* ca = atoms + (size_t)bi * 15 + 3;
    const float cax = ca[0], cay = ca[1], caz = ca[2];
    const float mi  = mask[bi];

    float lmax = 0.0f;
    for (int j = tid; j < LL; j += 256) {
        const float* p = atoms + (size_t)(base + j) * 15 + 3;
        float dx = cax - p[0], dy = cay - p[1], dz = caz - p[2];
        float m2 = mi * mask[base + j];
        float D  = m2 * sqrtf(dx*dx + dy*dy + dz*dz + 1e-6f);
        Dl[j] = D;
        lmax  = fmaxf(lmax, D);
    }
    red[tid] = lmax; __syncthreads();
    for (int s = 128; s > 0; s >>= 1) {
        if (tid < s) red[tid] = fmaxf(red[tid], red[tid + s]);
        __syncthreads();
    }
    const float Dmax = red[0];
    __syncthreads();
    for (int j = tid; j < LL; j += 256) {
        float m2 = mi * mask[base + j];
        Dl[j] += (1.0f - m2) * Dmax;       // D_adjust
    }
    __syncthreads();

    for (int it = 0; it < KNB; it++) {
        float bv = 3.0e38f; int bj = 0x7fffffff;
        for (int j = tid; j < LL; j += 256) {
            float v = Dl[j];
            if (v < bv || (v == bv && j < bj)) { bv = v; bj = j; }
        }
        red[tid] = bv; redi[tid] = bj; __syncthreads();
        for (int s = 128; s > 0; s >>= 1) {
            if (tid < s) {
                float ov = red[tid + s]; int oj = redi[tid + s];
                if (ov < red[tid] || (ov == red[tid] && oj < redi[tid])) {
                    red[tid] = ov; redi[tid] = oj;
                }
            }
            __syncthreads();
        }
        if (tid == 0) {
            int jm = redi[0];
            int o  = bi * KNB + it;
            eidx[o]       = jm;
            dnb[o]        = red[0];
            out_eidx_f[o] = (float)jm;
            Dl[jm] = 3.4e38f;              // remove from candidates
        }
        __syncthreads();
    }
}

// ---------------------------------------------------------------------------
// Kernel 4 (fused): feature build in LDS -> WMMA GEMM (416x128) -> LayerNorm.
// 64 edge rows / block, 256 threads = 8 waves. Wave w: row-tile w>>1, col-half w&1.
// ---------------------------------------------------------------------------
#define APAD  NF                               // halves per LDS A row (416, 16B-aligned stride)
#define ABYTES (ROWS_PER_BLK * APAD * 2)       // 53248
#define SMEM_BYTES (ABYTES + 512)              // + mu/rs overlay tail

__global__ void __launch_bounds__(256)
k_edge(const float* __restrict__ atoms, const int* __restrict__ eidx,
       const float* __restrict__ dnb, const _Float16* __restrict__ Wc,
       const int* __restrict__ ridx, const int* __restrict__ chains,
       const float* __restrict__ Wpos, const float* __restrict__ bpos,
       const float* __restrict__ gamma, const float* __restrict__ beta,
       float* __restrict__ out) {
    __shared__ __align__(16) char smem[SMEM_BYTES];
    _Float16* Alds = (_Float16*)smem;
    float*    Elds = (float*)smem;                 // overlay after GEMM (64*128*4 = 32KB < 52KB)
    float*    muld = (float*)(smem + ABYTES);      // 64 floats
    float*    rsld = muld + ROWS_PER_BLK;          // 64 floats

    const int tid = threadIdx.x;
    const int blk = blockIdx.x;

    // ---- Stage 1: build 64 rows x 416 f16 features; 4 threads per row -----
    {
        const int row = tid >> 2;                  // 0..63
        const int sub = tid & 3;
        const int R   = blk * ROWS_PER_BLK + row;  // global edge row
        const int b   = R / (LL * KNB);
        const int rem = R - b * (LL * KNB);
        const int i   = rem / KNB;
        const int j   = eidx[R];
        _Float16* f = Alds + row * APAD;

        float Aq[15], Nb[15];
        const float* pa = atoms + (size_t)(b * LL + i) * 15;
        const float* pb = atoms + (size_t)(b * LL + j) * 15;
        for (int a = 0; a < 15; a++) { Aq[a] = pa[a]; Nb[a] = pb[a]; }

        if (sub == 0) {
            // positional embedding
            int off  = ridx[b * LL + i] - ridx[b * LL + j];
            int same = (chains[b * LL + i] == chains[b * LL + j]);
            int d    = same ? min(max(off + 32, 0), 64) : 65;
            for (int m = 0; m < 16; m++) f[m] = (_Float16)(Wpos[d * 16 + m] + bpos[m]);
            // Ca-Ca RBF (of D_adjust selected value, matching reference)
            float Dn = dnb[R];
            for (int m = 0; m < 16; m++) f[16 + m] = (_Float16)rbf_val(Dn, m);
        }
        // 24 atom-pair RBF blocks, 6 pairs per sub-thread
        for (int q = 0; q < 6; q++) {
            int p  = sub + 4 * q;
            int ia = cPA[p] * 3, ib = cPB[p] * 3;
            float dx = Aq[ia] - Nb[ib], dy = Aq[ia+1] - Nb[ib+1], dz = Aq[ia+2] - Nb[ib+2];
            float D  = sqrtf(dx*dx + dy*dy + dz*dz + 1e-6f);
            _Float16* fp = f + 32 + p * 16;
            for (int m = 0; m < 16; m++) fp[m] = (_Float16)rbf_val(D, m);
        }
    }
    __syncthreads();

    // ---- Stage 2: WMMA GEMM: (64x416) x (416x128), f32 accumulate ---------
    const int w  = tid >> 5;          // wave 0..7
    const int l  = tid & 31;
    const int lr = l & 15;
    const int hi = (l >> 4) & 1;      // lane half
    const int rt = w >> 1;            // row-tile 0..3 (16 rows each)
    const int ch = w & 1;             // column half: cols [ch*64, ch*64+64)

    v8f acc[4] = { {}, {}, {}, {} };
    const int arow = rt * 16 + lr;
    for (int t = 0; t < 13; t++) {    // 13 K-tiles of 32
        // A fragment (ISA 16-bit A layout): lanes 0-15: K t*32+{0..7,16..23};
        // lanes 16-31: K t*32+{8..15,24..31}
        union { v16h v; v8h h[2]; } ua;
        const _Float16* ap = Alds + arow * APAD + t * 32 + hi * 8;
        ua.h[0] = *(const v8h*)(ap);
        ua.h[1] = *(const v8h*)(ap + 16);
        for (int ci = 0; ci < 4; ci++) {
            // B fragment: lane -> column ct*16+lr, K same as A fragment (col-major Wc)
            const int col = (ch * 4 + ci) * 16 + lr;
            union { v16h v; v8h h[2]; } ub;
            const _Float16* bp = Wc + col * NF + t * 32 + hi * 8;
            ub.h[0] = *(const v8h*)(bp);
            ub.h[1] = *(const v8h*)(bp + 16);
            acc[ci] = __builtin_amdgcn_wmma_f32_16x16x32_f16(
                false, ua.v, false, ub.v, (short)0, acc[ci], false, false);
        }
    }
    __syncthreads();                   // everyone done reading Alds

    // ---- Stage 3: spill D tiles to LDS (overlay), LayerNorm, store --------
    for (int ci = 0; ci < 4; ci++) {
        const int col = (ch * 4 + ci) * 16 + lr;
        for (int jj = 0; jj < 8; jj++) {
            int rloc = rt * 16 + jj + hi * 8;      // ISA C/D layout
            Elds[rloc * NOUT + col] = acc[ci][jj];
        }
    }
    __syncthreads();
    if (tid < ROWS_PER_BLK) {
        const float* e = Elds + tid * NOUT;
        float s = 0.0f, s2 = 0.0f;
        for (int c = 0; c < NOUT; c++) { float v = e[c]; s += v; s2 += v * v; }
        float mu  = s  * (1.0f / NOUT);
        float var = s2 * (1.0f / NOUT) - mu * mu;
        muld[tid] = mu;
        rsld[tid] = rsqrtf(var + 1e-5f);
    }
    __syncthreads();
    const size_t obase = (size_t)blk * (ROWS_PER_BLK * NOUT);
    for (int idx = tid; idx < ROWS_PER_BLK * NOUT; idx += 256) {
        int row = idx >> 7, col = idx & 127;
        out[obase + idx] = (Elds[idx] - muld[row]) * rsld[row] * gamma[col] + beta[col];
    }
}

// ---------------------------------------------------------------------------
// Launch
// ---------------------------------------------------------------------------
extern "C" void kernel_launch(void* const* d_in, const int* in_sizes, int n_in,
                              void* d_out, int out_size, void* d_ws, size_t ws_size,
                              hipStream_t stream) {
    const float* X      = (const float*)d_in[0];
    const float* mask   = (const float*)d_in[1];
    const int*   ridx   = (const int*)  d_in[2];
    const int*   chains = (const int*)  d_in[3];
    const float* Wpos   = (const float*)d_in[4];
    const float* bpos   = (const float*)d_in[5];
    const float* We     = (const float*)d_in[6];
    const float* gamma  = (const float*)d_in[7];
    const float* beta   = (const float*)d_in[8];
    float* out = (float*)d_out;

    char* ws = (char*)d_ws;
    float*     atoms = (float*)ws;                        // 8192*15*4   = 491,520 B
    int*       eidx  = (int*)  (ws + 491520);             // 262144*4    = 1,048,576 B
    float*     dnb   = (float*)(ws + 1540096);            // 262144*4    = 1,048,576 B
    _Float16*  Wc    = (_Float16*)(ws + 2588672);         // 53248*2     = 106,496 B

    float* out_eidx_f = out + (size_t)NROWS * NOUT;       // tail: E_idx as float

    k_atoms <<<(BB*LL + 255) / 256, 256, 0, stream>>>(X, atoms);
    k_wstage<<<(NF*NOUT + 255) / 256, 256, 0, stream>>>(We, Wc);
    k_topk  <<<BB*LL, 256, 0, stream>>>(atoms, mask, eidx, dnb, out_eidx_f);
    k_edge  <<<NROWS / ROWS_PER_BLK, 256, 0, stream>>>(
        atoms, eidx, dnb, Wc, ridx, chains, Wpos, bpos, gamma, beta, out);
}
